// minRNNLayer_26345329394158
// MI455X (gfx1250) — matrix operand order repaired and checked by reading, not compiled, not run on hardware
//
#include <hip/hip_runtime.h>
#include <hip/hip_bf16.h>
#include <stdint.h>

// Problem dims
#define BB 8
#define SSQ 4096
#define EE 512
#define HH 512
#define MT (BB * SSQ)          // 32768 tokens
#define LN_EPS 1e-5f

typedef __bf16 bf16_t;
typedef __attribute__((ext_vector_type(16))) __bf16 v16bf;
typedef __attribute__((ext_vector_type(8)))  __bf16 v8bf;
typedef __attribute__((ext_vector_type(8)))  float  v8f;

// ---------------------------------------------------------------- math utils
__device__ __forceinline__ float softplus_f(float x) {
  if (x > 20.f)  return x;
  if (x < -20.f) return __expf(x);
  return log1pf(__expf(x));
}
__device__ __forceinline__ float log_g_f(float x) {
  return (x >= 0.f) ? __logf(x + 0.5f) : -softplus_f(-x);
}
__device__ __forceinline__ float logaddexp_f(float a, float b) {
  float m = fmaxf(a, b), n = fminf(a, b);
  if (m == -INFINITY) return m;
  return m + log1pf(__expf(n - m));
}

// ----------------------------------------------------- CDNA5 async LDS copy
// GLOBAL_LOAD_ASYNC_TO_LDS_B128: per-lane 16B global -> LDS, tracked by ASYNCcnt.
__device__ __forceinline__ void async_load_b128(const bf16_t* gptr, bf16_t* lptr) {
  auto l3 = (__attribute__((address_space(3))) bf16_t*)lptr;
  asm volatile("global_load_async_to_lds_b128 %0, %1, off"
               :: "v"(l3), "v"(gptr) : "memory");
}
__device__ __forceinline__ void wait_async0() {
  asm volatile("s_wait_asynccnt 0x0" ::: "memory");
}

// ------------------------------------------------------- WMMA fragment loads
// A fragment: 16x32 bf16 (rows r0.., K=kb..kb+31), row-major src with stride ld.
// Lanes 0-15: row=lane, elems 0..7 = K0..7, elems 8..15 = K16..23
// Lanes 16-31: row=lane-16, elems 0..7 = K8..15, elems 8..15 = K24..31
__device__ __forceinline__ v16bf load_fragA(const bf16_t* __restrict__ base,
                                            int ld, int r0, int kb, int lane) {
  int r = lane & 15;
  int koff = (lane >> 4) << 3;             // 0 or 8
  const bf16_t* p = base + (size_t)(r0 + r) * (size_t)ld + (size_t)(kb + koff);
  union { v16bf f; v8bf h[2]; } u;
  u.h[0] = *(const v8bf*)p;
  u.h[1] = *(const v8bf*)(p + 16);
  return u.f;
}
// B fragment: 32x16 (KxN) from row-major weight W[n][k].
// Lanes 0-15: col=lane, elems 0..7 = K0..7, elems 8..15 = K16..23; lanes 16-31: K8..15, K24..31.
__device__ __forceinline__ v16bf load_fragB(const bf16_t* __restrict__ W,
                                            int ld, int n0, int kb, int lane) {
  int n = lane & 15;
  int kst = (lane >> 4) << 3;              // 0 or 8
  const bf16_t* p = W + (size_t)(n0 + n) * (size_t)ld + (size_t)(kb + kst);
  union { v16bf f; v8bf h[2]; } u;
  u.h[0] = *(const v8bf*)p;
  u.h[1] = *(const v8bf*)(p + 16);
  return u.f;
}

// ------------------------------------------------------------- weight packing
__global__ void __launch_bounds__(256) cvt_kernel(const float* __restrict__ src,
                                                  bf16_t* __restrict__ dst, int n) {
  int i = blockIdx.x * 256 + threadIdx.x;
  if (i < n) dst[i] = (bf16_t)src[i];
}
// conv_w (E_out, E_in, 4) -> dst[n][k*512 + ei]
__global__ void __launch_bounds__(256) convpack_kernel(const float* __restrict__ src,
                                                       bf16_t* __restrict__ dst) {
  int i = blockIdx.x * 256 + threadIdx.x;
  if (i < EE * EE * 4) {
    int n   = i >> 11;
    int rem = i & 2047;         // k*512 + ei
    int k   = rem >> 9;
    int ei  = rem & 511;
    dst[i] = (bf16_t)src[(size_t)n * 2048 + (size_t)ei * 4 + k];
  }
}

// ------------------------------------------------------------------ layernorm
__global__ void __launch_bounds__(256) ln_kernel(const float* __restrict__ x,
                                                 const float* __restrict__ g,
                                                 const float* __restrict__ bta,
                                                 float* __restrict__ y32,
                                                 bf16_t* __restrict__ y16) {
  int row = blockIdx.x;
  const float* xr = x + (size_t)row * EE;
  int tid = threadIdx.x;
  float v0 = xr[tid], v1 = xr[tid + 256];
  float s = v0 + v1, q = v0 * v0 + v1 * v1;
#pragma unroll
  for (int o = 16; o > 0; o >>= 1) {
    s += __shfl_xor(s, o, 32);
    q += __shfl_xor(q, o, 32);
  }
  __shared__ float ls[8], lq[8];
  int wid = tid >> 5, lane = tid & 31;
  if (lane == 0) { ls[wid] = s; lq[wid] = q; }
  __syncthreads();
  if (tid == 0) {
    float ts = 0.f, tq = 0.f;
#pragma unroll
    for (int i = 0; i < 8; i++) { ts += ls[i]; tq += lq[i]; }
    float mu  = ts / (float)EE;
    float var = tq / (float)EE - mu * mu;
    ls[0] = mu;
    lq[0] = rsqrtf(var + LN_EPS);
  }
  __syncthreads();
  float mu = ls[0], rs = lq[0];
  float o0 = (v0 - mu) * rs * g[tid] + bta[tid];
  float o1 = (v1 - mu) * rs * g[tid + 256] + bta[tid + 256];
  size_t base = (size_t)row * EE;
  y32[base + tid] = o0;
  y32[base + tid + 256] = o1;
  y16[base + tid] = (bf16_t)o0;
  y16[base + tid + 256] = (bf16_t)o1;
}

// --------------------------------------------- fused gate GEMM (f, i, h~ -> logs)
__global__ void __launch_bounds__(256) gate_gemm(const bf16_t* __restrict__ A,
                                                 const bf16_t* __restrict__ Wf,
                                                 const bf16_t* __restrict__ Wi,
                                                 const bf16_t* __restrict__ Wh,
                                                 const float* __restrict__ bf_,
                                                 const float* __restrict__ bi_,
                                                 const float* __restrict__ bh_,
                                                 float* __restrict__ p_lf,
                                                 float* __restrict__ p_lv) {
  int wave = blockIdx.x * 8 + (threadIdx.x >> 5);
  int lane = threadIdx.x & 31;
  int m0 = (wave >> 5) << 4;               // HH/16 = 32 tiles in N
  int n0 = (wave & 31) << 4;
  v8f accf = {}, acci = {}, acch = {};
  for (int kb = 0; kb < EE; kb += 32) {
    v16bf a  = load_fragA(A, EE, m0, kb, lane);
    v16bf wf = load_fragB(Wf, EE, n0, kb, lane);
    v16bf wi = load_fragB(Wi, EE, n0, kb, lane);
    v16bf wh = load_fragB(Wh, EE, n0, kb, lane);
    accf = __builtin_amdgcn_wmma_f32_16x16x32_bf16(false, a, false, wf, (short)0, accf, false, false);
    acci = __builtin_amdgcn_wmma_f32_16x16x32_bf16(false, a, false, wi, (short)0, acci, false, false);
    acch = __builtin_amdgcn_wmma_f32_16x16x32_bf16(false, a, false, wh, (short)0, acch, false, false);
  }
  int n  = n0 + (lane & 15);
  int mb = (lane >> 4) << 3;
  float bfv = bf_[n], biv = bi_[n], bhv = bh_[n];
#pragma unroll
  for (int i = 0; i < 8; i++) {
    int m = m0 + mb + i;
    float f  = accf[i] + bfv;
    float ii = acci[i] + biv;
    float hh = acch[i] + bhv;
    float diff  = softplus_f(-f) - softplus_f(-ii);
    float lf    = -softplus_f(diff);
    float li    = -softplus_f(-diff);
    size_t idx  = (size_t)m * HH + n;
    p_lf[idx] = lf;
    p_lv[idx] = li + log_g_f(hh);
  }
}

// ----------------------------------------------------------------- log scan
__global__ void __launch_bounds__(256) scan_kernel(const float* __restrict__ p_lf,
                                                   const float* __restrict__ p_lv,
                                                   const float* __restrict__ h0,
                                                   bf16_t* __restrict__ hseq,
                                                   float* __restrict__ hlast) {
  int c = blockIdx.x * 256 + threadIdx.x;  // 0 .. B*H-1
  int b = c >> 9;
  int h = c & 511;
  float lh = log_g_f(h0[c]);
  size_t idx = (size_t)b * SSQ * HH + h;
  for (int s = 0; s < SSQ; s++) {
    lh = logaddexp_f(p_lf[idx] + lh, p_lv[idx]);
    float hv = __expf(lh);
    hseq[idx] = (bf16_t)hv;
    if (s == SSQ - 1) hlast[c] = hv;
    idx += HH;
  }
}

// ------------------------------------------------------ LDS-staged async GEMM
// Block: 256 threads = 8 waves, tile M128 x N64, K-step 32. Wave: 32x32 (2x2 frags).
// A/B panels staged into LDS by global_load_async_to_lds_b128, double buffered.
// EPI 0: outf = acc + bias[n] + res  (f32);  EPI 1: outb = bf16(relu(acc + bias[n]))
template <int EPI>
__global__ void __launch_bounds__(256) gemm_lds(const bf16_t* __restrict__ A,
                                                const bf16_t* __restrict__ Bw,
                                                const float* __restrict__ bias,
                                                const float* __restrict__ res,
                                                float* __restrict__ outf,
                                                bf16_t* __restrict__ outb,
                                                int M, int N, int K) {
  __shared__ __align__(16) bf16_t As[2][128 * 32];   // 8 KB per buffer
  __shared__ __align__(16) bf16_t Bs[2][64 * 32];    // 4 KB per buffer
  int tid  = threadIdx.x;
  int wave = tid >> 5, lane = tid & 31;
  int tilesN = N >> 6;
  int m0 = (blockIdx.x / tilesN) << 7;
  int n0 = (blockIdx.x % tilesN) << 6;
  int wm = wave >> 1, wn = wave & 1;

  // stage one K-panel (A: 128x32, B: 64x32) into LDS buffer sb, fully async
  auto stage = [&](int kb, int sb) {
    // A: 512 x 16B chunks, 2 per thread
    {
      int c0 = tid;               // chunk 0
      int row = c0 >> 2, cc = c0 & 3;
      async_load_b128(A + (size_t)(m0 + row) * K + kb + cc * 8,
                      &As[sb][row * 32 + cc * 8]);
      int c1 = tid + 256;         // chunk 1
      row = c1 >> 2; cc = c1 & 3;
      async_load_b128(A + (size_t)(m0 + row) * K + kb + cc * 8,
                      &As[sb][row * 32 + cc * 8]);
    }
    // B: 256 x 16B chunks, 1 per thread
    {
      int row = tid >> 2, cc = tid & 3;
      async_load_b128(Bw + (size_t)(n0 + row) * K + kb + cc * 8,
                      &Bs[sb][row * 32 + cc * 8]);
    }
  };

  stage(0, 0);
  wait_async0();
  __syncthreads();

  v8f acc[2][2] = {};
  int buf = 0;
  for (int kb = 0; kb < K; kb += 32) {
    if (kb + 32 < K) stage(kb + 32, buf ^ 1);   // overlap DMA with compute

    v16bf af[2], bfr[2];
#pragma unroll
    for (int mf = 0; mf < 2; mf++) {
      int row  = wm * 32 + mf * 16 + (lane & 15);
      int koff = (lane >> 4) << 3;
      const bf16_t* p = &As[buf][row * 32 + koff];
      union { v16bf f; v8bf h[2]; } u;
      u.h[0] = *(const v8bf*)p;
      u.h[1] = *(const v8bf*)(p + 16);
      af[mf] = u.f;
    }
#pragma unroll
    for (int nf = 0; nf < 2; nf++) {
      int row = wn * 32 + nf * 16 + (lane & 15);
      int kst = (lane >> 4) << 3;
      const bf16_t* p = &Bs[buf][row * 32 + kst];
      union { v16bf f; v8bf h[2]; } u;
      u.h[0] = *(const v8bf*)p;
      u.h[1] = *(const v8bf*)(p + 16);
      bfr[nf] = u.f;
    }
#pragma unroll
    for (int mf = 0; mf < 2; mf++)
#pragma unroll
      for (int nf = 0; nf < 2; nf++)
        acc[mf][nf] = __builtin_amdgcn_wmma_f32_16x16x32_bf16(
            false, af[mf], false, bfr[nf], (short)0, acc[mf][nf], false, false);

    wait_async0();       // next-stage DMA done (had the whole compute to land)
    __syncthreads();     // all waves' panels visible / reads complete
    buf ^= 1;
  }

  // epilogue: wave writes its 32x32 block
  int mbase = m0 + wm * 32;
  int nbase = n0 + wn * 32;
#pragma unroll
  for (int nf = 0; nf < 2; nf++) {
    int n = nbase + nf * 16 + (lane & 15);
    float bv = bias[n];
#pragma unroll
    for (int mf = 0; mf < 2; mf++) {
      int mrow = mbase + mf * 16 + ((lane >> 4) << 3);
#pragma unroll
      for (int i = 0; i < 8; i++) {
        size_t idx = (size_t)(mrow + i) * N + n;
        float v = acc[mf][nf][i] + bv;
        if (EPI == 0) outf[idx] = v + res[idx];
        else          outb[idx] = (bf16_t)fmaxf(v, 0.f);
      }
    }
  }
}

// -------------------------------------------- conv as GEMM (K = 4*512, shifted A)
__global__ void __launch_bounds__(256) conv_gemm(const bf16_t* __restrict__ A,   // xn2 bf16 [MT][EE]
                                                 const bf16_t* __restrict__ W,   // [EE][4*EE] packed
                                                 const float* __restrict__ bias,
                                                 const float* __restrict__ res,
                                                 float* __restrict__ outf) {
  int wave = blockIdx.x * 8 + (threadIdx.x >> 5);
  int lane = threadIdx.x & 31;
  int m0 = (wave >> 5) << 4;               // EE/16 = 32 N-tiles
  int n0 = (wave & 31) << 4;
  int r  = lane & 15;
  int koff = (lane >> 4) << 3;
  int m = m0 + r;
  int b = m >> 12;                         // / 4096
  int s = m & 4095;
  v8f acc = {};
  for (int kb = 0; kb < 4 * EE; kb += 32) {
    int kid = kb >> 9;                     // kernel tap 0..3
    int ei  = (kb & 511) + koff;
    int srow = s + kid - 1;                // padding (1,2)
    union { v16bf f; v8bf h[2]; uint32_t z[8]; } u;
#pragma unroll
    for (int i = 0; i < 8; i++) u.z[i] = 0u;
    if (srow >= 0 && srow < SSQ) {
      const bf16_t* p = A + ((size_t)(b << 12) + srow) * EE + ei;
      u.h[0] = *(const v8bf*)p;
      u.h[1] = *(const v8bf*)(p + 16);
    }
    v16bf w = load_fragB(W, 4 * EE, n0, kb, lane);
    acc = __builtin_amdgcn_wmma_f32_16x16x32_bf16(false, u.f, false, w, (short)0, acc, false, false);
  }
  int n  = n0 + (lane & 15);
  int mb = (lane >> 4) << 3;
  float bv = bias[n];
#pragma unroll
  for (int i = 0; i < 8; i++) {
    int mm = m0 + mb + i;
    size_t idx = (size_t)mm * EE + n;
    outf[idx] = acc[i] + bv + res[idx];
  }
}

// ------------------------------------------------------------------- launcher
extern "C" void kernel_launch(void* const* d_in, const int* in_sizes, int n_in,
                              void* d_out, int out_size, void* d_ws, size_t ws_size,
                              hipStream_t stream) {
  (void)in_sizes; (void)n_in; (void)out_size; (void)ws_size;
  const float* x      = (const float*)d_in[0];
  const float* h0     = (const float*)d_in[1];
  const float* lf_w   = (const float*)d_in[2];
  const float* lf_b   = (const float*)d_in[3];
  const float* li_w   = (const float*)d_in[4];
  const float* li_b   = (const float*)d_in[5];
  const float* lh_w   = (const float*)d_in[6];
  const float* lh_b   = (const float*)d_in[7];
  const float* out_w  = (const float*)d_in[8];
  const float* out_b  = (const float*)d_in[9];
  const float* conv_w = (const float*)d_in[10];
  const float* conv_b = (const float*)d_in[11];
  const float* ln_g   = (const float*)d_in[12];
  const float* ln_b   = (const float*)d_in[13];
  const float* w1     = (const float*)d_in[14];
  const float* b1     = (const float*)d_in[15];
  const float* w2     = (const float*)d_in[16];
  const float* b2     = (const float*)d_in[17];
  float* out = (float*)d_out;

  char* ws = (char*)d_ws;
  const size_t ACT = (size_t)MT * EE;      // 16,777,216 elements
  float*  bufA = (float*)ws;                                   // 64 MB
  float*  bufB = (float*)(ws + ACT * 4);                       // 64 MB
  float*  bufC = (float*)(ws + ACT * 8);                       // 64 MB
  bf16_t* bufN = (bf16_t*)(ws + ACT * 12);                     // 32 MB act bf16
  bf16_t* bufM = (bf16_t*)(ws + ACT * 14);                     // 128 MB hmid bf16
  bf16_t* bufW = (bf16_t*)(ws + ACT * 14 + (size_t)MT * 2048 * 2);
  bf16_t* Wf = bufW;
  bf16_t* Wi = Wf + 512 * 512;
  bf16_t* Wh = Wi + 512 * 512;
  bf16_t* Wo = Wh + 512 * 512;
  bf16_t* W1 = Wo + 512 * 512;
  bf16_t* W2 = W1 + 2048 * 512;
  bf16_t* Wc = W2 + 512 * 2048;

  // 1) pack weights to bf16
  cvt_kernel<<<(512 * 512 + 255) / 256, 256, 0, stream>>>(lf_w, Wf, 512 * 512);
  cvt_kernel<<<(512 * 512 + 255) / 256, 256, 0, stream>>>(li_w, Wi, 512 * 512);
  cvt_kernel<<<(512 * 512 + 255) / 256, 256, 0, stream>>>(lh_w, Wh, 512 * 512);
  cvt_kernel<<<(512 * 512 + 255) / 256, 256, 0, stream>>>(out_w, Wo, 512 * 512);
  cvt_kernel<<<(2048 * 512 + 255) / 256, 256, 0, stream>>>(w1, W1, 2048 * 512);
  cvt_kernel<<<(512 * 2048 + 255) / 256, 256, 0, stream>>>(w2, W2, 512 * 2048);
  convpack_kernel<<<(EE * EE * 4 + 255) / 256, 256, 0, stream>>>(conv_w, Wc);

  // 2) LN1: x -> xn1 (bufA f32, bufN bf16)
  ln_kernel<<<MT, 256, 0, stream>>>(x, ln_g, ln_b, bufA, bufN);

  // 3) fused gate GEMM -> log_f (bufB), log_v (bufC)
  gate_gemm<<<(MT / 16) * (HH / 16) / 8, 256, 0, stream>>>(
      bufN, Wf, Wi, Wh, lf_b, li_b, lh_b, bufB, bufC);

  // 4) scan -> h_seq bf16 (bufN, xn1 bf16 now dead) + h_last f32 tail of d_out
  scan_kernel<<<(BB * HH) / 256, 256, 0, stream>>>(
      bufB, bufC, h0, bufN, out + (size_t)BB * SSQ * EE);

  // 5) out-proj GEMM: x2 = xn1 + hseq @ Wo^T + out_b  -> bufB
  gemm_lds<0><<<(MT / 128) * (512 / 64), 256, 0, stream>>>(
      bufN, Wo, out_b, bufA, bufB, nullptr, MT, 512, 512);

  // 6) LN2: bufB -> xn2 (bufC f32, bufN bf16)
  ln_kernel<<<MT, 256, 0, stream>>>(bufB, ln_g, ln_b, bufC, bufN);

  // 7) conv GEMM: x3 = xn2 + conv(xn2) + conv_b -> bufB
  conv_gemm<<<(MT / 16) * (EE / 16) / 8, 256, 0, stream>>>(
      bufN, Wc, conv_b, bufC, bufB);

  // 8) LN3: bufB -> xn3 (bufA f32, bufN bf16)
  ln_kernel<<<MT, 256, 0, stream>>>(bufB, ln_g, ln_b, bufA, bufN);

  // 9) MLP1: hmid = relu(xn3 @ W1^T + b1) -> bufM (bf16)
  gemm_lds<1><<<(MT / 128) * (2048 / 64), 256, 0, stream>>>(
      bufN, W1, b1, nullptr, nullptr, bufM, MT, 2048, 512);

  // 10) MLP2: out = xn3 + hmid @ W2^T + b2 -> d_out (f32)
  gemm_lds<0><<<(MT / 128) * (512 / 64), 256, 0, stream>>>(
      bufM, W2, b2, bufA, out, nullptr, MT, 512, 2048);
}
